// Model_20547123544450
// MI455X (gfx1250) — matrix operand order repaired and checked by reading, not compiled, not run on hardware
//
#include <hip/hip_runtime.h>
#include <hip/hip_bf16.h>

typedef __bf16 bf16;
typedef __bf16 v16bf __attribute__((ext_vector_type(16)));
typedef __bf16 v8bf  __attribute__((ext_vector_type(8)));
typedef __bf16 v2bf  __attribute__((ext_vector_type(2)));
typedef float  v8f   __attribute__((ext_vector_type(8)));
typedef float  v4f   __attribute__((ext_vector_type(4)));

#define HW   3136
#define WID  56

// ---------------------------------------------------------------------------
// 1) fp32 -> bf16 convert (x), 8 elements per thread (2x b128 in, b128 out)
// ---------------------------------------------------------------------------
__global__ __launch_bounds__(256) void cvt_f32_bf16(const float* __restrict__ src,
                                                    bf16* __restrict__ dst, int n8) {
    int i = blockIdx.x * 256 + threadIdx.x;
    if (i >= n8) return;
    v4f f0 = *(const v4f*)(src + (size_t)i * 8);
    v4f f1 = *(const v4f*)(src + (size_t)i * 8 + 4);
    v8bf o;
    o[0] = (bf16)f0.x; o[1] = (bf16)f0.y; o[2] = (bf16)f0.z; o[3] = (bf16)f0.w;
    o[4] = (bf16)f1.x; o[5] = (bf16)f1.y; o[6] = (bf16)f1.z; o[7] = (bf16)f1.w;
    *(v8bf*)(dst + (size_t)i * 8) = o;
}

// ---------------------------------------------------------------------------
// 2) pack weights into WMMA A-fragment layout (bf16), zero-padded.
//    dst flat index = (((g*mtiles + mt)*ksteps + ks)*32 + lane)*16 + e
//    lane = (m&15) + 16*half ; k32 = e + 16*half
// ---------------------------------------------------------------------------
__global__ __launch_bounds__(256) void pack_wfrags(const float* __restrict__ src,
                                                   bf16* __restrict__ dst,
                                                   int rows_real, int mtiles,
                                                   int K_real, int ksteps, int total) {
    int tid = blockIdx.x * 256 + threadIdx.x;
    if (tid >= total) return;
    int e    = tid & 15;
    int lane = (tid >> 4) & 31;
    int rest = tid >> 9;
    int ks = rest % ksteps; rest /= ksteps;
    int mt = rest % mtiles;
    int g  = rest / mtiles;
    int half = lane >> 4;
    int m = mt * 16 + (lane & 15);
    int k = ks * 32 + e + 16 * half;
    float v = 0.0f;
    if (m < rows_real && k < K_real)
        v = src[((size_t)(g * rows_real + m)) * K_real + k];
    dst[tid] = (bf16)v;
}

// ---------------------------------------------------------------------------
// Stage a K=32 x N=64 bf16 B-tile into LDS in WMMA B-fragment order:
//   slot(nb, L, e) = lds[((nb*32 + L)<<4) | e] holds X[k = e + 16*(L>>4),
//                                                     p = nb*16 + (L&15)]
// Each slot covers 2 K-rows x 8 pixels: 2x global_load_b128 +
// 8x packed b32 LDS stores (compiler merges into ds_store_2addr_b32).
// FULL=true: all 32 channels valid -> no EXEC-mask guards in the hot path.
// ---------------------------------------------------------------------------
template <bool FULL>
__device__ __forceinline__ void stage_b_tile(bf16* lds, const bf16* __restrict__ base,
                                             int cbase, int climit, int hw0,
                                             int tid, int nthreads) {
    for (int idx = tid; idx < 128; idx += nthreads) {
        int k0 = (idx >> 3) << 1;      // 0,2,...,30
        int j0 = (idx & 7) << 3;       // 0,8,...,56
        v8bf r0 = {}, r1 = {};
        int c0 = cbase + k0;
        const bf16* p0 = base + (size_t)c0 * HW + hw0 + j0;
        if (FULL || c0 < climit)     r0 = *(const v8bf*)(p0);
        if (FULL || c0 + 1 < climit) r1 = *(const v8bf*)(p0 + HW);
        int Lb = (k0 >> 4) << 4;
        int e0 = k0 & 15;              // even
#pragma unroll
        for (int jj = 0; jj < 8; ++jj) {
            int j = j0 + jj;
            int L = Lb | (j & 15);
            v2bf p; p.x = r0[jj]; p.y = r1[jj];
            *(v2bf*)(lds + ((((j >> 4) * 32 + L) << 4) | e0)) = p;
        }
    }
}

// consume staged tile: 1 A-fragment (2x b128) + 4 B-fragments + 4 WMMAs
__device__ __forceinline__ void wmma4(v8f acc[4], const bf16* __restrict__ afrag,
                                      const bf16* lds, int lane) {
    v16bf a = *(const v16bf*)afrag;
#pragma unroll
    for (int nb = 0; nb < 4; ++nb) {
        v16bf b = *(const v16bf*)(lds + ((nb * 32 + lane) << 4));
        acc[nb] = __builtin_amdgcn_wmma_f32_16x16x32_bf16(
            false, a, false, b, (short)0, acc[nb], false, false);
    }
}

// ---------------------------------------------------------------------------
// 3) conv1: grouped 1x1 GEMM (M=40->48 pad per group, K=80->96 pad), BN+ReLU,
//    output t1 bf16 [n][120][hw].  Block = 96 threads (3 waves, 3 M-tiles).
// ---------------------------------------------------------------------------
__global__ __launch_bounds__(96) void conv1_gemm(const bf16* __restrict__ x16,
                                                 const bf16* __restrict__ w1p,
                                                 const float* __restrict__ scale1,
                                                 const float* __restrict__ shift1,
                                                 bf16* __restrict__ t1) {
    __shared__ __align__(32) bf16 lds[2048];
    int bx  = blockIdx.x;            // 0..1567 : n * 49 + ntile
    int n   = bx / 49;
    int hw0 = (bx % 49) * 64;
    int g   = blockIdx.y;            // 0..2
    int tid  = threadIdx.x;
    int wave = tid >> 5;             // 0..2  -> M tile in group
    int lane = tid & 31;

    const bf16* xn = x16 + (size_t)n * 240 * HW;
    const bf16* wp = w1p + ((((size_t)g * 3 + wave) * 3) * 32 + lane) * 16;
    v8f acc[4] = {};

    // steps 0,1: all 32 channels valid; step 2: channels 64..79 valid
    for (int ks = 0; ks < 2; ++ks) {
        __syncthreads();
        stage_b_tile<true>(lds, xn, g * 80 + ks * 32, 0, hw0, tid, 96);
        __syncthreads();
        wmma4(acc, wp + (size_t)ks * 512, lds, lane);
    }
    __syncthreads();
    stage_b_tile<false>(lds, xn, g * 80 + 64, g * 80 + 80, hw0, tid, 96);
    __syncthreads();
    wmma4(acc, wp + (size_t)2 * 512, lds, lane);

    int half = lane >> 4, col = lane & 15;
#pragma unroll
    for (int r = 0; r < 8; ++r) {
        int mloc = wave * 16 + r + 8 * half;     // 0..47 virtual rows
        if (mloc < 40) {
            int c = g * 40 + mloc;
            float sc = scale1[c], sh = shift1[c];
            bf16* dst = t1 + ((size_t)(n * 120 + c)) * HW + hw0 + col;
#pragma unroll
            for (int nb = 0; nb < 4; ++nb) {
                float y = fmaxf(acc[nb][r] * sc + sh, 0.0f);
                dst[nb * 16] = (bf16)y;
            }
        }
    }
}

// ---------------------------------------------------------------------------
// 4) depthwise 3x3 + BN2 + channel shuffle (g=3): t1 -> t2 (both bf16)
// ---------------------------------------------------------------------------
__global__ __launch_bounds__(256) void dwconv3x3(const bf16* __restrict__ t1,
                                                 const float* __restrict__ w2,
                                                 const float* __restrict__ scale2,
                                                 const float* __restrict__ shift2,
                                                 bf16* __restrict__ t2) {
    int c = blockIdx.y;              // 0..119
    int n = blockIdx.z;              // 0..31
    int hw = blockIdx.x * 256 + threadIdx.x;
    if (hw >= HW) return;
    int h = hw / WID, w = hw % WID;
    const bf16* p = t1 + ((size_t)(n * 120 + c)) * HW;
    const float* wk = w2 + c * 9;
    float acc = 0.0f;
#pragma unroll
    for (int dy = -1; dy <= 1; ++dy) {
#pragma unroll
        for (int dx = -1; dx <= 1; ++dx) {
            int hh = h + dy, ww = w + dx;
            if (hh >= 0 && hh < WID && ww >= 0 && ww < WID)
                acc += (float)p[hh * WID + ww] * wk[(dy + 1) * 3 + (dx + 1)];
        }
    }
    acc = acc * scale2[c] + shift2[c];
    int co = (c % 40) * 3 + (c / 40);   // channel shuffle
    t2[((size_t)(n * 120 + co)) * HW + hw] = (bf16)acc;
}

// ---------------------------------------------------------------------------
// 5) final GEMM: shortcut (480x240 over x) + conv3 (grouped, 40K over t2),
//    fused epilogue relu(BN3) + BN_sc -> out fp32.
//    Block = 160 threads (5 waves = 5 M-tiles of one group half).
// ---------------------------------------------------------------------------
__global__ __launch_bounds__(160) void final_gemm(const bf16* __restrict__ x16,
                                                  const bf16* __restrict__ t2,
                                                  const bf16* __restrict__ wscp,
                                                  const bf16* __restrict__ w3p,
                                                  const float* __restrict__ scale3,
                                                  const float* __restrict__ shift3,
                                                  const float* __restrict__ scale_sc,
                                                  const float* __restrict__ shift_sc,
                                                  float* __restrict__ out) {
    __shared__ __align__(32) bf16 lds[2048];
    int bx  = blockIdx.x;             // 0..1567
    int n   = bx / 49;
    int hw0 = (bx % 49) * 64;
    int by  = blockIdx.y;             // 0..5
    int g   = by >> 1;
    int tid  = threadIdx.x;
    int wave = tid >> 5;              // 0..4
    int lane = tid & 31;
    int mtg  = (by & 1) * 5 + wave;   // M tile within group, 0..9
    int m0   = g * 160 + mtg * 16;
    int mtile_global = g * 10 + mtg;  // 0..29

    const bf16* xn  = x16 + (size_t)n * 240 * HW;
    const bf16* t2n = t2  + (size_t)n * 120 * HW;
    const bf16* wpS = wscp + ((size_t)mtile_global * 8 * 32 + lane) * 16;
    const bf16* wp3 = w3p  + ((size_t)mtile_global * 2 * 32 + lane) * 16;

    v8f accS[4] = {};
    v8f acc3[4] = {};

    // ---- shortcut: K = 240 padded to 256; steps 0..6 full, step 7 partial ----
    for (int ks = 0; ks < 7; ++ks) {
        __builtin_prefetch(xn + (size_t)((ks + 1) * 32) * HW + hw0, 0, 0);
        __syncthreads();
        stage_b_tile<true>(lds, xn, ks * 32, 0, hw0, tid, 160);
        __syncthreads();
        wmma4(accS, wpS + (size_t)ks * 512, lds, lane);
    }
    __syncthreads();
    stage_b_tile<false>(lds, xn, 224, 240, hw0, tid, 160);
    __syncthreads();
    wmma4(accS, wpS + (size_t)7 * 512, lds, lane);

    // ---- conv3: group K = 40 padded to 64; step 0 full, step 1 partial ----
    __syncthreads();
    stage_b_tile<true>(lds, t2n, g * 40, 0, hw0, tid, 160);
    __syncthreads();
    wmma4(acc3, wp3, lds, lane);
    __syncthreads();
    stage_b_tile<false>(lds, t2n, g * 40 + 32, g * 40 + 40, hw0, tid, 160);
    __syncthreads();
    wmma4(acc3, wp3 + 512, lds, lane);

    // ---- fused epilogue: relu(BN3(y)) + BN_sc(z) ----
    int half = lane >> 4, col = lane & 15;
    float s3[8], h3[8], ss[8], hs[8];
#pragma unroll
    for (int r = 0; r < 8; ++r) {
        int m = m0 + r + 8 * half;
        s3[r] = scale3[m];   h3[r] = shift3[m];
        ss[r] = scale_sc[m]; hs[r] = shift_sc[m];
    }
    float* on = out + (size_t)n * 480 * HW;
#pragma unroll
    for (int r = 0; r < 8; ++r) {
        int m = m0 + r + 8 * half;
        float* dst = on + (size_t)m * HW + hw0 + col;
#pragma unroll
        for (int nb = 0; nb < 4; ++nb) {
            float y = fmaxf(acc3[nb][r] * s3[r] + h3[r], 0.0f);
            y += accS[nb][r] * ss[r] + hs[r];
            dst[nb * 16] = y;
        }
    }
}

// ---------------------------------------------------------------------------
extern "C" void kernel_launch(void* const* d_in, const int* in_sizes, int n_in,
                              void* d_out, int out_size, void* d_ws, size_t ws_size,
                              hipStream_t stream) {
    (void)in_sizes; (void)n_in; (void)out_size; (void)ws_size;

    const float* x        = (const float*)d_in[0];
    const float* w1       = (const float*)d_in[1];
    const float* scale1   = (const float*)d_in[2];
    const float* shift1   = (const float*)d_in[3];
    const float* w2       = (const float*)d_in[4];
    const float* scale2   = (const float*)d_in[5];
    const float* shift2   = (const float*)d_in[6];
    const float* w3       = (const float*)d_in[7];
    const float* scale3   = (const float*)d_in[8];
    const float* shift3   = (const float*)d_in[9];
    const float* wsc      = (const float*)d_in[10];
    const float* scale_sc = (const float*)d_in[11];
    const float* shift_sc = (const float*)d_in[12];
    float* out = (float*)d_out;

    // workspace layout (all 256B aligned)
    const size_t szX  = (size_t)32 * 240 * HW * sizeof(bf16);   // 48,168,960
    const size_t szT  = (size_t)32 * 120 * HW * sizeof(bf16);   // 24,084,480
    char* ws = (char*)d_ws;
    bf16* X16  = (bf16*)(ws);
    bf16* T1   = (bf16*)(ws + szX);
    bf16* T2   = (bf16*)(ws + szX + szT);
    bf16* W1P  = (bf16*)(ws + szX + 2 * szT);                 // 13,824 el
    bf16* W3P  = (bf16*)(ws + szX + 2 * szT + 32768);         // 30,720 el
    bf16* WSCP = (bf16*)(ws + szX + 2 * szT + 32768 + 65536); // 122,880 el

    // 1) convert x to bf16
    {
        int n8 = (32 * 240 * HW) / 8;           // 3,010,560
        cvt_f32_bf16<<<(n8 + 255) / 256, 256, 0, stream>>>(x, X16, n8);
    }
    // 2) pack weight fragments
    {
        int tot1 = 3 * 3 * 3 * 512;     // w1:  g=3, mtiles=3 (40->48), ksteps=3 (80->96)
        pack_wfrags<<<(tot1 + 255) / 256, 256, 0, stream>>>(w1, W1P, 40, 3, 80, 3, tot1);
        int tot3 = 3 * 10 * 2 * 512;    // w3:  g=3, mtiles=10 (160), ksteps=2 (40->64)
        pack_wfrags<<<(tot3 + 255) / 256, 256, 0, stream>>>(w3, W3P, 160, 10, 40, 2, tot3);
        int totS = 1 * 30 * 8 * 512;    // wsc: g=1, mtiles=30 (480), ksteps=8 (240->256)
        pack_wfrags<<<(totS + 255) / 256, 256, 0, stream>>>(wsc, WSCP, 480, 30, 240, 8, totS);
    }
    // 3) conv1 grouped 1x1 GEMM + BN + ReLU -> T1
    conv1_gemm<<<dim3(32 * 49, 3), 96, 0, stream>>>(X16, W1P, scale1, shift1, T1);
    // 4) depthwise 3x3 + BN + shuffle -> T2
    dwconv3x3<<<dim3((HW + 255) / 256, 120, 32), 256, 0, stream>>>(T1, w2, scale2, shift2, T2);
    // 5) shortcut GEMM + conv3 GEMM + fused epilogue -> out
    final_gemm<<<dim3(32 * 49, 6), 160, 0, stream>>>(X16, T2, WSCP, W3P,
                                                     scale3, shift3, scale_sc, shift_sc, out);
}
